// SAE_8315056685706
// MI455X (gfx1250) — matrix operand order repaired and checked by reading, not compiled, not run on hardware
//
#include <hip/hip_runtime.h>

typedef __attribute__((ext_vector_type(16))) __bf16 v16bf;
typedef __attribute__((ext_vector_type(8)))  float  v8f;
typedef __attribute__((ext_vector_type(4)))  float  v4f;
typedef __attribute__((ext_vector_type(4)))  unsigned int v4u;
typedef __attribute__((ext_vector_type(8)))  int    v8i;
typedef __attribute__((ext_vector_type(4)))  int    v4i;

#define B_ROWS 4096
#define D_IN   1024
#define HIDDEN 32768
#define TOPK   32

// ---------------------------------------------------------------------------
// Prep 1: W_enc f32 [HIDDEN, D_IN] -> bf16, pre-swizzled into WMMA B-fragment
// layout: Wswz[ht][kk][lane][e].  B (K=32 x N=16, 16-bit): lanes 0-15 hold
// K={0..7,16..23} (n = lane), lanes 16-31 hold K={8..15,24..31}; 16 bf16 per lane.
// ---------------------------------------------------------------------------
__global__ void sae_swizzle_wenc(const float* __restrict__ W_enc,
                                 __bf16* __restrict__ Wswz) {
  const size_t total = (size_t)HIDDEN * D_IN;
  for (size_t i = (size_t)blockIdx.x * blockDim.x + threadIdx.x; i < total;
       i += (size_t)gridDim.x * blockDim.x) {
    unsigned h  = (unsigned)(i >> 10);
    unsigned d  = (unsigned)(i & 1023u);
    unsigned ht = h >> 4, n = h & 15u;
    unsigned kk = d >> 5, kl = d & 31u;
    unsigned half = (kl >> 3) & 1u, e = (kl & 7u) | (((kl >> 4) & 1u) << 3);
    unsigned lane = (half << 4) | n;
    size_t slot = (((size_t)ht * 32 + kk) * 32 + lane) * 16 + e;
    Wswz[slot] = (__bf16)W_enc[i];
  }
}

// ---------------------------------------------------------------------------
// Prep 2: (x - b_dec) f32 -> bf16, pre-swizzled into WMMA A-fragment layout
// per ISA 7.12.2 (16-bit A 16x32): lanes 0-15: M=lane, K={0..7,16..23};
// lanes 16-31: M=lane-16, K={8..15,24..31}.  Xswz[rb][kk][lane][e].
// ---------------------------------------------------------------------------
__global__ void sae_swizzle_x(const float* __restrict__ x,
                              const float* __restrict__ b_dec,
                              __bf16* __restrict__ Xswz) {
  const size_t total = (size_t)B_ROWS * D_IN;
  for (size_t i = (size_t)blockIdx.x * blockDim.x + threadIdx.x; i < total;
       i += (size_t)gridDim.x * blockDim.x) {
    unsigned row = (unsigned)(i >> 10);
    unsigned d   = (unsigned)(i & 1023u);
    unsigned rb = row >> 4, m = row & 15u;
    unsigned kk = d >> 5, kl = d & 31u;
    unsigned half = (kl >> 3) & 1u;
    unsigned e = (kl & 7u) | (((kl >> 4) & 1u) << 3);
    unsigned lane = (half << 4) | m;
    size_t slot = (((size_t)rb * 32 + kk) * 32 + lane) * 16 + e;
    Xswz[slot] = (__bf16)(x[i] - b_dec[d]);
  }
}

// ---------------------------------------------------------------------------
// Fused encoder-GEMM (bf16 WMMA, dual accumulators) + running top-32 + decode.
// Grid: B_ROWS/16 blocks x 256 threads (8 wave32s). Each block owns 16 rows
// and sweeps all 32768 hidden cols; W_enc bf16 (67MB) stays in the 192MB L2.
// A-tile (32KB) is DMA'd into LDS once via the Tensor Data Mover.
// ---------------------------------------------------------------------------
__global__ __launch_bounds__(256) void sae_fused(
    const __bf16* __restrict__ Xswz, const __bf16* __restrict__ Wswz,
    const float* __restrict__ b_enc, const float* __restrict__ W_dec,
    const float* __restrict__ b_dec, float* __restrict__ out) {
  __shared__ __align__(32) __bf16 aTile[32 * 32 * 16];  // 32 KB: 16 rows x K=1024
  __shared__ float stage[16 * 256];                     // 16 KB: pre_acts tile
  __shared__ float cand[256];                           // per-subblock maxima
  __shared__ float tkv[16][TOPK];                       // running top-k values
  __shared__ int   tki[16][TOPK];                       // running top-k indices

  const int tid  = threadIdx.x;
  const int lane = tid & 31;
  const int wave = tid >> 5;
  const int rb   = blockIdx.x;  // 16-row block

  // ---- one-shot A-tile fill: TDM global->LDS (32KB contiguous), wave 0 ----
#if __has_builtin(__builtin_amdgcn_tensor_load_to_lds)
  if (wave == 0) {
    const unsigned long long gaddr =
        (unsigned long long)(uintptr_t)(Xswz) + (unsigned long long)rb * 32768ull;
    const unsigned laddr = (unsigned)(uintptr_t)(&aTile[0]);  // low32 = LDS offset
    v4u g0 = {0, 0, 0, 0};
    g0[0] = 1u;                           // count = 1 valid descriptor
    g0[1] = laddr;                        // lds_addr [63:32]
    g0[2] = (unsigned)(gaddr & 0xffffffffull);          // global_addr [95:64]
    g0[3] = (unsigned)((gaddr >> 32) & 0x1ffffffull)    // global_addr [120:96]
            | 0x80000000u;                              // type = 2 ("image")
    v8i g1 = {0, 0, 0, 0, 0, 0, 0, 0};
    g1[0] = 1 << 16;                      // data_size = 1 (2 bytes/elem)
    g1[1] = 16384 << 16;                  // tensor_dim0 = 16384 (lo16 @ [63:48])
    g1[2] = 1 << 16;                      // tensor_dim0 hi=0; tensor_dim1 = 1
    g1[3] = 16384 << 16;                  // tile_dim0 = 16384 @ [127:112]
    g1[4] = 1;                            // tile_dim1 = 1
    g1[5] = 16384;                        // tensor_dim0_stride = 16384
    v4i gz = {0, 0, 0, 0};
#if __has_include(<hip/amd_detail/amd_gfx1250_TDM.h>)
    v8i gz8 = {0, 0, 0, 0, 0, 0, 0, 0};
    __builtin_amdgcn_tensor_load_to_lds(g0, g1, gz, gz, gz8, 0);  // clang-23 form
#else
    __builtin_amdgcn_tensor_load_to_lds(g0, g1, gz, gz, 0);       // ROCm 7.2 form
#endif
    __builtin_amdgcn_s_wait_tensorcnt(0);
  }
#else
  {
    const uint4* src = (const uint4*)(Xswz) + (size_t)rb * 2048;
    uint4* dst = (uint4*)aTile;
#pragma unroll
    for (int i = 0; i < 8; ++i) dst[tid + i * 256] = src[tid + i * 256];
  }
#endif
  for (int i = tid; i < 16 * TOPK; i += 256) {
    (&tkv[0][0])[i] = -1.0e30f;
    (&tki[0][0])[i] = 0;
  }
  float curmin = -1.0e30f;  // per-row threshold (manager threads tid<16)
  int   amin   = 0;
  __syncthreads();

  const v16bf* aFrag = (const v16bf*)aTile;

  for (int it = 0; it < HIDDEN / 256; ++it) {
    // --- GEMM: wave computes 16 rows x 32 hidden cols (2 tiles), K=1024 ---
    const int ht0 = it * 16 + wave * 2;
    const v16bf* bFrag = (const v16bf*)Wswz + (size_t)ht0 * 1024;
    __builtin_prefetch(bFrag + 2048, 0, 1);  // start pulling next slab (L2->L0)
    v8f acc0 = {};
    v8f acc1 = {};
#pragma unroll 4
    for (int kk = 0; kk < 32; ++kk) {
      v16bf a  = aFrag[kk * 32 + lane];         // 2x ds_load_b128, shared by both
      v16bf b0 = bFrag[kk * 32 + lane];         // 2x global_load_b128 (L2)
      v16bf b1 = bFrag[1024 + kk * 32 + lane];
      acc0 = __builtin_amdgcn_wmma_f32_16x16x32_bf16(
          false, a, false, b0, (short)0, acc0, false, false);
      acc1 = __builtin_amdgcn_wmma_f32_16x16x32_bf16(
          false, a, false, b1, (short)0, acc1, false, false);
    }

    __syncthreads();  // previous iteration's top-k scan is done with `stage`
    {
      // C layout: lanes 0-15 -> M=0..7 in VGPR 0..7; lanes 16-31 -> M=8..15.
      const int col   = lane & 15;
      const int rbase = (lane >> 4) * 8;
      const int c0 = wave * 32 + col;           // stage col of tile 0
      const float bias0 = b_enc[it * 256 + c0];
      const float bias1 = b_enc[it * 256 + c0 + 16];
#pragma unroll
      for (int v = 0; v < 8; ++v) {
        stage[(rbase + v) * 256 + c0]      = fmaxf(acc0[v] + bias0, 0.0f);
        stage[(rbase + v) * 256 + c0 + 16] = fmaxf(acc1[v] + bias1, 0.0f);
      }
    }
    __syncthreads();
    {
      // 16 threads per row each reduce 16 candidates -> threshold gate.
      const int row = tid >> 4, sub = tid & 15;
      const float* s = &stage[row * 256 + sub * 16];
      float m = s[0];
#pragma unroll
      for (int j = 1; j < 16; ++j) m = fmaxf(m, s[j]);
      cand[tid] = m;
    }
    __syncthreads();
    if (tid < 16) {  // one manager thread per row
      float gm = cand[tid * 16];
#pragma unroll
      for (int j = 1; j < 16; ++j) gm = fmaxf(gm, cand[tid * 16 + j]);
      if (gm > curmin) {  // rare after warm-up (~k*ln(H/k) inserts total)
        const float* s = &stage[tid * 256];
        for (int c = 0; c < 256; ++c) {
          float v = s[c];
          if (v > curmin) {
            tkv[tid][amin] = v;
            tki[tid][amin] = it * 256 + c;
            curmin = tkv[tid][0];
            amin   = 0;
            for (int j = 1; j < TOPK; ++j) {
              float tv = tkv[tid][j];
              if (tv < curmin) { curmin = tv; amin = j; }
            }
          }
        }
      }
    }
  }
  __syncthreads();

  // --- sparse decode: x_rec[r] = sum_j tkv[r][j] * W_dec[tki[r][j]] + b_dec ---
  {
    const int row = tid >> 4, l16 = tid & 15;
    float* orow = out + ((size_t)rb * 16 + row) * D_IN;
    for (int c = 0; c < 16; ++c) {
      const int d0 = c * 64 + l16 * 4;  // 16 lanes -> 256B coalesced chunks
      v4f accd = *(const v4f*)(b_dec + d0);
#pragma unroll 8
      for (int j = 0; j < TOPK; ++j) {
        const float v = tkv[row][j];
        const v4f w = *(const v4f*)(W_dec + (size_t)tki[row][j] * D_IN + d0);
        accd += v * w;
      }
      *(v4f*)(orow + d0) = accd;
    }
  }
}

extern "C" void kernel_launch(void* const* d_in, const int* in_sizes, int n_in,
                              void* d_out, int out_size, void* d_ws, size_t ws_size,
                              hipStream_t stream) {
  const float* x     = (const float*)d_in[0];
  const float* W_enc = (const float*)d_in[1];
  const float* b_enc = (const float*)d_in[2];
  const float* W_dec = (const float*)d_in[3];
  const float* b_dec = (const float*)d_in[4];
  float* out = (float*)d_out;

  // Workspace: [Wswz bf16: 64MB][Xswz bf16: 8MB]
  __bf16* Wswz = (__bf16*)d_ws;
  __bf16* Xswz = (__bf16*)((char*)d_ws + (size_t)HIDDEN * D_IN * sizeof(__bf16));

  sae_swizzle_wenc<<<8192, 256, 0, stream>>>(W_enc, Wswz);
  sae_swizzle_x<<<1024, 256, 0, stream>>>(x, b_dec, Xswz);
  sae_fused<<<B_ROWS / 16, 256, 0, stream>>>(Xswz, Wswz, b_enc, W_dec, b_dec, out);
}
